// coRNN_30167850287887
// MI455X (gfx1250) — matrix operand order repaired and checked by reading, not compiled, not run on hardware
//
#include <hip/hip_runtime.h>

// coRNN on MI455X (gfx1250): bf16 WMMA GEMMs + persistent wave-register scan.
//   T=1024, B=64, N_INP=128, N_HID=512, N_OUT=128, dt=0.01, gamma=eps=1

#define TSTEPS 1024
#define BATCH  64
#define NINP   128
#define NHID   512
#define NOUT   128
#define DT_C   0.01f
#define NWG    8            // workgroups in persistent scan kernel
#define HPW    (NHID / NWG) // 64 hidden rows owned per workgroup
#define LDP    (NHID + 8)   // padded LDS k-stride (bank-conflict-free column reads)
#define LDPI   (NINP + 8)

typedef __attribute__((ext_vector_type(16))) __bf16 v16bf;
typedef __attribute__((ext_vector_type(8)))  __bf16 v8bf;
typedef __attribute__((ext_vector_type(8)))  float  v8f;

// D = A(16x32 bf16) x B(32x16 bf16) + C(16x16 f32), wave32 WMMA
#define WMMA_BF16(a, b, c) \
  __builtin_amdgcn_wmma_f32_16x16x32_bf16(false, (a), false, (b), (short)0, (c), false, false)

// A fragment (16x32), converting f32 -> bf16 on the fly.
// Lane l: row m=l&15; lanes 0..15 hold K {0..7,16..23}, lanes 16..31 hold K {8..15,24..31}.
static __device__ inline v16bf load_a_f32(const float* __restrict__ p0, int ld, int lane) {
  const int m = lane & 15, h8 = (lane >> 4) << 3;
  const float* p = p0 + (size_t)m * ld;
  v16bf a;
#pragma unroll
  for (int j = 0; j < 8; ++j) {
    a[j]     = (__bf16)p[h8 + j];
    a[j + 8] = (__bf16)p[16 + h8 + j];
  }
  return a;
}

// A fragment (16x32) from packed bf16 (two 16B vector loads per lane).
static __device__ inline v16bf load_a_bf16(const __bf16* __restrict__ p0, int ld, int lane) {
  const int m = lane & 15, h8 = (lane >> 4) << 3;
  const __bf16* p = p0 + (size_t)m * ld;
  v8bf lo = *(const v8bf*)(p + h8);
  v8bf hi = *(const v8bf*)(p + 16 + h8);
  v16bf a;
#pragma unroll
  for (int j = 0; j < 8; ++j) { a[j] = lo[j]; a[j + 8] = hi[j]; }
  return a;
}

// B fragment (32x16): lane l covers column n=l&15; lanes 0..15 hold K {0..7,16..23}, lanes 16..31 hold K {8..15,24..31}.
// Source is LDS weight slice stored [n][k] row-major with padded stride ld.
static __device__ inline v16bf load_b_lds(const __bf16* p0, int ld, int lane) {
  const int n = lane & 15, kh = (lane >> 4) << 3;
  const __bf16* p = p0 + n * ld + kh;
  v8bf lo = *(const v8bf*)(p);
  v8bf hi = *(const v8bf*)(p + 16);
  v16bf b;
#pragma unroll
  for (int j = 0; j < 8; ++j) { b[j] = lo[j]; b[j + 8] = hi[j]; }
  return b;
}

__global__ void zero_ws(unsigned* w, int n) {
  int i = blockIdx.x * blockDim.x + threadIdx.x;
  if (i < n) w[i] = 0u;
}

// Phase 1: i_ext[tb,h] = x[tb,:] @ W_I[h,:] + b_I[h]  (written into the save_act slot)
__global__ __launch_bounds__(512) void iext_kernel(const float* __restrict__ x,
                                                   const float* __restrict__ Wi,
                                                   const float* __restrict__ bi,
                                                   float* __restrict__ iext) {
  __shared__ __bf16 lW[HPW * LDPI];
  const int tid = threadIdx.x;
  const int M0 = blockIdx.x * 64;   // flat (t*B+b) row tile
  const int H0 = blockIdx.y * 64;   // hidden column tile
#pragma unroll 4
  for (int i = tid; i < HPW * NINP; i += 512) {
    int h = i >> 7, k = i & (NINP - 1);
    lW[h * LDPI + k] = (__bf16)Wi[(size_t)(H0 + h) * NINP + k];
  }
  __syncthreads();
  const int lane = tid & 31, wave = tid >> 5;
  const int tM = wave & 3, tN = wave >> 2;
  const int n15 = lane & 15, mb = (lane >> 4) << 3;
  const float bias = bi[H0 + tN * 16 + n15];
  v8f c0, c1;
#pragma unroll
  for (int r = 0; r < 8; ++r) { c0[r] = bias; c1[r] = 0.f; }
#pragma unroll
  for (int kk = 0; kk < NINP / 64; ++kk) {   // two independent accumulation chains
    const int k0 = kk * 64;
    v16bf a0 = load_a_f32(x + (size_t)(M0 + tM * 16) * NINP + k0, NINP, lane);
    v16bf b0 = load_b_lds(lW + (tN * 16) * LDPI + k0, LDPI, lane);
    c0 = WMMA_BF16(a0, b0, c0);
    v16bf a1 = load_a_f32(x + (size_t)(M0 + tM * 16) * NINP + k0 + 32, NINP, lane);
    v16bf b1 = load_b_lds(lW + (tN * 16) * LDPI + k0 + 32, LDPI, lane);
    c1 = WMMA_BF16(a1, b1, c1);
  }
  const int hcol = H0 + tN * 16 + n15;
#pragma unroll
  for (int r = 0; r < 8; ++r)
    iext[(size_t)(M0 + tM * 16 + mb + r) * NHID + hcol] = c0[r] + c1[r];
}

// Phase 2: persistent scan. 8 WGs; WG g owns hidden rows [64g,64g+64).
// Weights resident in LDS (bf16); f32 hy/hz state resident in wave registers;
// double-buffered bf16 hy/hz exchanged via L2 each step; atomic-counter barrier.
// W_R and W_F products use independent accumulators -> two interleavable WMMA chains.
__global__ __launch_bounds__(512) void scan_kernel(const float* __restrict__ Wr,
                                                   const float* __restrict__ Wf,
                                                   float* __restrict__ out_hy,
                                                   float* __restrict__ out_hz,
                                                   float* __restrict__ act_io,
                                                   __bf16* __restrict__ hyb,
                                                   __bf16* __restrict__ hzb,
                                                   unsigned* __restrict__ counter) {
  __shared__ __bf16 lWr[HPW * LDP];
  __shared__ __bf16 lWf[HPW * LDP];
  const int tid = threadIdx.x;
  const int H0 = blockIdx.x * HPW;
#pragma unroll 4
  for (int i = tid; i < HPW * NHID; i += 512) {
    int h = i >> 9, k = i & (NHID - 1);
    lWr[h * LDP + k] = (__bf16)Wr[(size_t)(H0 + h) * NHID + k];
    lWf[h * LDP + k] = (__bf16)Wf[(size_t)(H0 + h) * NHID + k];
  }
  __syncthreads();
  const int lane = tid & 31, wave = tid >> 5;
  const int tM = wave & 3, tN = wave >> 2;
  const int B0 = tM * 16;
  const int n15 = lane & 15, mb = (lane >> 4) << 3;
  const int hcol = H0 + tN * 16 + n15;      // hidden index owned by this lane
  const __bf16* lWrT = lWr + (tN * 16) * LDP;
  const __bf16* lWfT = lWf + (tN * 16) * LDP;
  float hy[8], hz[8];
#pragma unroll
  for (int r = 0; r < 8; ++r) { hy[r] = 0.f; hz[r] = 0.f; }

  for (int t = 0; t < TSTEPS; ++t) {
    float* actt = act_io + (size_t)t * (BATCH * NHID);
    v8f c0, c1;
#pragma unroll
    for (int r = 0; r < 8; ++r) {          // c0 starts at i_ext[t]; c1 independent chain
      c0[r] = actt[(B0 + mb + r) * NHID + hcol];
      c1[r] = 0.f;
    }
    const __bf16* hyr = hyb + (size_t)(t & 1) * (BATCH * NHID) + B0 * NHID;
    const __bf16* hzr = hzb + (size_t)(t & 1) * (BATCH * NHID) + B0 * NHID;
#pragma unroll
    for (int kk = 0; kk < NHID / 32; ++kk) {
      const int k0 = kk * 32;
      v16bf ay = load_a_bf16(hyr + k0, NHID, lane);
      v16bf br = load_b_lds(lWrT + k0, LDP, lane);
      c0 = WMMA_BF16(ay, br, c0);
      v16bf az = load_a_bf16(hzr + k0, NHID, lane);
      v16bf bf = load_b_lds(lWfT + k0, LDP, lane);
      c1 = WMMA_BF16(az, bf, c1);
    }
    if (t + 1 < TSTEPS)                    // global_prefetch_b8 for next step's i_ext
      __builtin_prefetch(actt + BATCH * NHID + (B0 + mb) * NHID + hcol, 0, 0);
    __bf16* hyw = hyb + (size_t)((t + 1) & 1) * (BATCH * NHID);
    __bf16* hzw = hzb + (size_t)((t + 1) & 1) * (BATCH * NHID);
#pragma unroll
    for (int r = 0; r < 8; ++r) {
      const int b = B0 + mb + r;
      const float a = tanhf(c0[r] + c1[r]);
      actt[b * NHID + hcol] = a;           // overwrite i_ext with act[t]
      const float z = hz[r] + DT_C * (a - hy[r] - hz[r]);
      const float y = hy[r] + DT_C * z;
      hz[r] = z; hy[r] = y;
      const size_t gi = (size_t)(t * BATCH + b) * NHID + hcol;
      out_hy[gi] = y;
      out_hz[gi] = z;
      hyw[b * NHID + hcol] = (__bf16)y;    // publish bf16 state for t+1
      hzw[b * NHID + hcol] = (__bf16)z;
    }
    // device-wide step barrier (release/acquire through L2)
    __threadfence();
    __syncthreads();
    if (tid == 0) {
      __hip_atomic_fetch_add(counter, 1u, __ATOMIC_RELEASE, __HIP_MEMORY_SCOPE_AGENT);
      const unsigned target = (unsigned)(t + 1) * NWG;
      while (__hip_atomic_load(counter, __ATOMIC_ACQUIRE, __HIP_MEMORY_SCOPE_AGENT) < target)
        __builtin_amdgcn_s_sleep(2);
    }
    __syncthreads();
    __threadfence();
  }
}

// Phase 3: out[tb,o] = hy[tb,:] @ W_ro[o,:] + b_ro[o]
__global__ __launch_bounds__(512) void readout_kernel(const float* __restrict__ hy,
                                                      const float* __restrict__ Wro,
                                                      const float* __restrict__ bro,
                                                      float* __restrict__ out) {
  __shared__ __bf16 lW[HPW * LDP];        // 64 output rows x 512 K
  const int tid = threadIdx.x;
  const int M0 = blockIdx.x * 64;
  const int O0 = blockIdx.y * 64;
#pragma unroll 4
  for (int i = tid; i < HPW * NHID; i += 512) {
    int o = i >> 9, k = i & (NHID - 1);
    lW[o * LDP + k] = (__bf16)Wro[(size_t)(O0 + o) * NHID + k];
  }
  __syncthreads();
  const int lane = tid & 31, wave = tid >> 5;
  const int tM = wave & 3, tN = wave >> 2;
  const int n15 = lane & 15, mb = (lane >> 4) << 3;
  const float bias = bro[O0 + tN * 16 + n15];
  v8f c0, c1;
#pragma unroll
  for (int r = 0; r < 8; ++r) { c0[r] = bias; c1[r] = 0.f; }
#pragma unroll 2
  for (int kk = 0; kk < NHID / 64; ++kk) { // two independent accumulation chains
    const int k0 = kk * 64;
    v16bf a0 = load_a_f32(hy + (size_t)(M0 + tM * 16) * NHID + k0, NHID, lane);
    v16bf b0 = load_b_lds(lW + (tN * 16) * LDP + k0, LDP, lane);
    c0 = WMMA_BF16(a0, b0, c0);
    v16bf a1 = load_a_f32(hy + (size_t)(M0 + tM * 16) * NHID + k0 + 32, NHID, lane);
    v16bf b1 = load_b_lds(lW + (tN * 16) * LDP + k0 + 32, LDP, lane);
    c1 = WMMA_BF16(a1, b1, c1);
  }
  const int ocol = O0 + tN * 16 + n15;
#pragma unroll
  for (int r = 0; r < 8; ++r)
    out[(size_t)(M0 + tM * 16 + mb + r) * NOUT + ocol] = c0[r] + c1[r];
}

extern "C" void kernel_launch(void* const* d_in, const int* in_sizes, int n_in,
                              void* d_out, int out_size, void* d_ws, size_t ws_size,
                              hipStream_t stream) {
  (void)in_sizes; (void)n_in; (void)out_size; (void)ws_size;
  const float* x   = (const float*)d_in[0];
  const float* Wi  = (const float*)d_in[1];
  const float* bi  = (const float*)d_in[2];
  const float* Wr  = (const float*)d_in[3];
  const float* Wf  = (const float*)d_in[4];
  const float* Wro = (const float*)d_in[5];
  const float* bro = (const float*)d_in[6];

  float* out     = (float*)d_out;
  float* out_y   = out;                                          // [T,B,NOUT]
  float* out_hy  = out + (size_t)TSTEPS * BATCH * NOUT;          // [T,B,NHID]
  float* out_hz  = out_hy + (size_t)TSTEPS * BATCH * NHID;       // [T,B,NHID]
  float* out_act = out_hz + (size_t)TSTEPS * BATCH * NHID;       // [T,B,NHID] (i_ext then act)

  // Workspace: 2x double-buffered bf16 state (4 x 64KB) + barrier counter.
  __bf16* hyb = (__bf16*)d_ws;
  __bf16* hzb = hyb + 2 * BATCH * NHID;
  unsigned* counter = (unsigned*)((char*)d_ws + 4 * (size_t)BATCH * NHID * sizeof(__bf16));
  const int zn = (int)(4 * BATCH * NHID * sizeof(__bf16) / 4) + 32;
  zero_ws<<<(zn + 255) / 256, 256, 0, stream>>>((unsigned*)d_ws, zn);

  iext_kernel<<<dim3((TSTEPS * BATCH) / 64, NHID / 64), 512, 0, stream>>>(x, Wi, bi, out_act);
  scan_kernel<<<dim3(NWG), 512, 0, stream>>>(Wr, Wf, out_hy, out_hz, out_act, hyb, hzb, counter);
  readout_kernel<<<dim3((TSTEPS * BATCH) / 64, NOUT / 64), 512, 0, stream>>>(out_hy, Wro, bro, out_y);
}
